// RetinaNet_66675072303612
// MI455X (gfx1250) — matrix operand order repaired and checked by reading, not compiled, not run on hardware
//
#include <hip/hip_runtime.h>
#include <hip/hip_bf16.h>

// ---------------------------------------------------------------------------
// RetinaNet head for MI455X (gfx1250): implicit-GEMM conv via bf16 WMMA with
// zero-halo NHWC features (branch-free inner loop, 6-way A-fragment reuse),
// then decode + per-image top-k / greedy NMS.
// ---------------------------------------------------------------------------

typedef __attribute__((ext_vector_type(16))) __bf16 v16bf;
typedef __attribute__((ext_vector_type(8)))  __bf16 v8bf;
typedef __attribute__((ext_vector_type(8)))  float  v8f;

#define NLEV          5
#define CIN           256
#define KTOT          2304      // 9 * 256
#define NANCH         32736
#define BIMG          16
#define PAD_PER_IMG   1528832   // sum_l (s+2)*(s+2)*256
#define MT_PER_IMG    341       // sum_l pos_l/16
#define WROWS         96        // 32 (reg pad) + 64 (cls pad)

__constant__ int c_s[NLEV]      = {64, 32, 16, 8, 4};
__constant__ int c_mt[NLEV]     = {256, 64, 16, 4, 1};
__constant__ int c_padOff[NLEV] = {0, 1115136, 1411072, 1494016, 1519616};
__constant__ int c_jOff[NLEV]   = {0, 24576, 30720, 32256, 32640};

__device__ __forceinline__ unsigned short f2bf(float f) {
    union { float f; unsigned int u; } v; v.f = f;
    unsigned int r = v.u + 0x7FFFu + ((v.u >> 16) & 1u);  // RNE
    return (unsigned short)(r >> 16);
}

// ---------------------------------------------------------------------------
// K0a: zero the padded feature buffer (halo must be 0)
// ---------------------------------------------------------------------------
__global__ void zero_feat(unsigned short* __restrict__ dst, int total) {
    int idx = blockIdx.x * 256 + threadIdx.x;
    if (idx < total) dst[idx] = 0;
}

// ---------------------------------------------------------------------------
// K0b: prepack conv weights OIHW f32 -> bf16 [level][row][k], k = tap*256+c.
// rows 0..31 = reg (24 real, padded 0), rows 32..95 = cls (54 real, padded 0)
// ---------------------------------------------------------------------------
__global__ void prepack_weights(const float* __restrict__ reg_w,
                                const float* __restrict__ cls_w,
                                unsigned short* __restrict__ Wt) {
    int idx = blockIdx.x * 256 + threadIdx.x;
    const int TOT = NLEV * WROWS * KTOT;
    if (idx >= TOT) return;
    int l   = idx / (WROWS * KTOT);
    int r   = idx % (WROWS * KTOT);
    int row = r / KTOT;
    int k   = r % KTOT;
    int g = k / CIN, c = k % CIN;
    int ky = g / 3, kx = g % 3;
    float val = 0.0f;
    if (row < 32) {
        if (row < 24)
            val = reg_w[(((size_t)(l * 24 + row) * CIN + c) * 3 + ky) * 3 + kx];
    } else {
        int o = row - 32;
        if (o < 54)
            val = cls_w[(((size_t)(l * 54 + o) * CIN + c) * 3 + ky) * 3 + kx];
    }
    Wt[idx] = f2bf(val);
}

// ---------------------------------------------------------------------------
// K1: feats NCHW f32 -> padded NHWC bf16 interior (halo stays 0)
// dst is already offset to this level's padded block.
// ---------------------------------------------------------------------------
__global__ void convert_feat(const float* __restrict__ src,
                             unsigned short* __restrict__ dst,
                             int s, int total) {
    int idx = blockIdx.x * 256 + threadIdx.x;
    if (idx >= total) return;
    int hw = s * s;
    int b = idx / (CIN * hw);
    int r = idx % (CIN * hw);
    int c = r / hw;
    int p = r % hw;
    int y = p / s, x = p % s;
    int sp = s + 2;
    dst[(size_t)b * PAD_PER_IMG + (size_t)((y + 1) * sp + (x + 1)) * CIN + c] = f2bf(src[idx]);
}

// ---------------------------------------------------------------------------
// K2: implicit-GEMM 3x3 conv via v_wmma_f32_16x16x32_bf16.
// One wave per 16-position M-tile; computes ALL 6 output N-tiles
// (2 reg + 4 cls) with A-fragment reuse.  K = 9 taps * 8 chunks of 32.
// ---------------------------------------------------------------------------
union AFrag { v16bf v; v8bf h[2]; };

__global__ __launch_bounds__(256) void conv_gemm(
        const unsigned short* __restrict__ featB,   // padded NHWC bf16
        const unsigned short* __restrict__ Wt,
        const float* __restrict__ regBias,
        const float* __restrict__ clsBias,
        float* __restrict__ delta,   // [B][NANCH][4]
        float* __restrict__ conf) {  // [B][NANCH][9]
    const int wave = blockIdx.x * (blockDim.x >> 5) + (threadIdx.x >> 5);
    const int lane = threadIdx.x & 31;

    int b  = wave / MT_PER_IMG;
    int rr = wave % MT_PER_IMG;
    int l = 0;
    while (l < NLEV - 1 && rr >= c_mt[l]) { rr -= c_mt[l]; ++l; }
    const int s     = c_s[l];
    const int sp    = s + 2;
    const int mtile = rr;

    const int m = lane & 15;
    const int h = lane >> 4;
    const int p = mtile * 16 + m;
    const int y = p / s, x = p % s;

    const unsigned short* fimg = featB + (size_t)b * PAD_PER_IMG + c_padOff[l];

    // 6 weight-row base pointers (lane-dependent): tiles 0,1 = reg; 2..5 = cls
    const unsigned short* wptr[6];
#pragma unroll
    for (int t = 0; t < 6; ++t) {
        const int wrow = t * 16 + (lane & 15);      // rows 0..95
        wptr[t] = Wt + ((size_t)l * WROWS + wrow) * KTOT + (size_t)h * 16;
    }

    v8f acc[6];
#pragma unroll
    for (int t = 0; t < 6; ++t) acc[t] = (v8f){};

    for (int g = 0; g < 9; ++g) {
        const int gy = g / 3, gx = g % 3;
        // padded coords: interior (y+1,x+1), tap offset (gy-1,gx-1)
        const unsigned short* abase = fimg + (size_t)((y + gy) * sp + (x + gx)) * CIN;
        if (g < 8) __builtin_prefetch(wptr[0] + (g + 1) * CIN, 0, 1);
#pragma unroll
        for (int cb = 0; cb < 8; ++cb) {
            const int c0 = cb * 32;
            AFrag a;
            a.h[0] = *(const v8bf*)(abase + c0 + h * 16);      // K = h*16 .. +7
            a.h[1] = *(const v8bf*)(abase + c0 + h * 16 + 8);  // K = h*16+8 .. +15
#pragma unroll
            for (int t = 0; t < 6; ++t) {
                const v16bf bw = *(const v16bf*)(wptr[t] + g * CIN + c0);
                acc[t] = __builtin_amdgcn_wmma_f32_16x16x32_bf16(
                        false, a.v, false, bw, (short)0, acc[t], false, false);
            }
        }
    }

    // Scatter C: lane n = lane&15 fixed out-channel, rows m = h*8 + r.
    const int n = lane & 15;
#pragma unroll
    for (int t = 0; t < 6; ++t) {
        const int head  = (t < 2) ? 0 : 1;
        const int ntile = head ? (t - 2) : t;
        const int o     = ntile * 16 + n;
        const int omax  = head ? 54 : 24;
        if (o < omax) {
            const int a6 = o % 6, ch = o / 6;
            const float bias = head ? clsBias[l * 54 + o] : regBias[l * 24 + o];
            const int jbase = c_jOff[l] + a6 * s * s;
#pragma unroll
            for (int r = 0; r < 8; ++r) {
                const int pp = mtile * 16 + h * 8 + r;
                const int j  = jbase + pp;
                const float v = acc[t][r] + bias;
                if (head == 0) delta[(((size_t)b * NANCH + j) << 2) + ch] = v;
                else           conf [((size_t)b * NANCH + j) * 9 + ch]    = v;
            }
        }
    }
}

// ---------------------------------------------------------------------------
// K3: decode boxes + softmax scores
// ---------------------------------------------------------------------------
__global__ void decode_kernel(const float* __restrict__ delta,
                              const float* __restrict__ conf,
                              const float* __restrict__ anchors,
                              float* __restrict__ boxes,   // [B][NANCH][4]
                              float* __restrict__ sArr,    // [B][NANCH]
                              int*   __restrict__ catArr) {
    int i = blockIdx.x * 256 + threadIdx.x;
    if (i >= BIMG * NANCH) return;
    const int j = i % NANCH;

    const float* dd = delta + (size_t)i * 4;
    const float* an = anchors + (size_t)j * 4;
    const float acx = an[0], acy = an[1], aw = an[2], ah = an[3];
    const float cx = dd[0] * 0.1f * aw + acx;
    const float cy = dd[1] * 0.1f * ah + acy;
    const float w  = __expf(dd[2] * 0.2f) * aw;
    const float hh = __expf(dd[3] * 0.2f) * ah;
    float* ob = boxes + (size_t)i * 4;
    ob[0] = cx - 0.5f * w;  ob[1] = cy - 0.5f * hh;
    ob[2] = cx + 0.5f * w;  ob[3] = cy + 0.5f * hh;

    const float* cf = conf + (size_t)i * 9;
    float mx = cf[0]; int cat = 0;
#pragma unroll
    for (int c = 1; c < 9; ++c) if (cf[c] > mx) { mx = cf[c]; cat = c; }
    float Z = 0.0f;
#pragma unroll
    for (int c = 0; c < 9; ++c) Z += __expf(cf[c] - mx);
    const float score = 1.0f / Z;                    // max softmax prob
    const bool valid = (score > 0.05f) && (cat != 0);
    sArr[i]   = valid ? score : -1.0f;
    catArr[i] = cat;
}

// ---------------------------------------------------------------------------
// K4: per-image top-1000 -> greedy class-offset NMS -> top-200 -> outputs
// ---------------------------------------------------------------------------
#define PRE_K 1000
#define OUT_K 200

__global__ __launch_bounds__(1024) void nms_kernel(
        const float* __restrict__ boxes,
        float* __restrict__ sArr,            // mutated (rewritten by K3 each call)
        const int* __restrict__ catArr,
        float* __restrict__ out) {
    __shared__ float redV[1024];
    __shared__ int   redI[1024];
    __shared__ float topS[PRE_K];
    __shared__ int   topIdx[PRE_K];
    __shared__ float topBox[PRE_K * 4];
    __shared__ int   topCat[PRE_K];
    __shared__ int   keep[PRE_K];
    __shared__ float keptS[PRE_K];

    const int b = blockIdx.x;
    const int t = threadIdx.x;
    float* sc = sArr + (size_t)b * NANCH;

    // ---- phase A: top-1000 (iterative argmax, low-index tie-break) ----
    for (int k = 0; k < PRE_K; ++k) {
        float best = -1e30f; int bi = NANCH;
        for (int i = t; i < NANCH; i += 1024) {
            float v = sc[i];
            if (v > best) { best = v; bi = i; }
        }
        redV[t] = best; redI[t] = bi;
        __syncthreads();
        for (int stp = 512; stp > 0; stp >>= 1) {
            if (t < stp) {
                float v2 = redV[t + stp]; int i2 = redI[t + stp];
                if (v2 > redV[t] || (v2 == redV[t] && i2 < redI[t])) {
                    redV[t] = v2; redI[t] = i2;
                }
            }
            __syncthreads();
        }
        if (t == 0) { topS[k] = redV[0]; topIdx[k] = redI[0]; sc[redI[0]] = -1e30f; }
        __syncthreads();
    }

    // ---- gather selected boxes / cats, init keep ----
    for (int i = t; i < PRE_K; i += 1024) {
        int idx = topIdx[i];
        const float* bx = boxes + ((size_t)b * NANCH + idx) * 4;
        topBox[i * 4 + 0] = bx[0]; topBox[i * 4 + 1] = bx[1];
        topBox[i * 4 + 2] = bx[2]; topBox[i * 4 + 3] = bx[3];
        topCat[i] = catArr[(size_t)b * NANCH + idx];
        keep[i]   = (topS[i] > 0.0f) ? 1 : 0;
    }
    __syncthreads();

    // ---- greedy NMS with class-offset boxes ----
    for (int i = 0; i < PRE_K - 1; ++i) {
        if (keep[i]) {
            const float ci = (float)topCat[i] * 4.0f;
            const float x0 = topBox[i * 4 + 0] + ci, y0 = topBox[i * 4 + 1] + ci;
            const float x1 = topBox[i * 4 + 2] + ci, y1 = topBox[i * 4 + 3] + ci;
            const float ai = (x1 - x0) * (y1 - y0);
            for (int j = i + 1 + t; j < PRE_K; j += 1024) {
                if (keep[j]) {
                    const float cj = (float)topCat[j] * 4.0f;
                    const float u0 = topBox[j * 4 + 0] + cj, v0 = topBox[j * 4 + 1] + cj;
                    const float u1 = topBox[j * 4 + 2] + cj, v1 = topBox[j * 4 + 3] + cj;
                    const float lx = fmaxf(x0, u0), ly = fmaxf(y0, v0);
                    const float rx = fminf(x1, u1), ry = fminf(y1, v1);
                    const float iw = fmaxf(rx - lx, 0.0f), ih = fmaxf(ry - ly, 0.0f);
                    const float inter = iw * ih;
                    const float aj = (u1 - u0) * (v1 - v0);
                    const float iou = inter / (ai + aj - inter + 1e-9f);
                    if (iou > 0.5f) keep[j] = 0;
                }
            }
        }
        __syncthreads();
    }

    // ---- phase C: top-200 of kept scores, emit outputs ----
    for (int i = t; i < PRE_K; i += 1024) keptS[i] = keep[i] ? topS[i] : -1.0f;
    __syncthreads();

    float* outBoxes  = out;                          // [16][200][4]
    float* outCats   = out + BIMG * OUT_K * 4;       // [16][200]
    float* outScores = outCats + BIMG * OUT_K;
    float* outValid  = outScores + BIMG * OUT_K;

    for (int k = 0; k < OUT_K; ++k) {
        float best = -1e30f; int bi = PRE_K;
        for (int i = t; i < PRE_K; i += 1024) {
            float v = keptS[i];
            if (v > best || (v == best && i < bi)) { best = v; bi = i; }
        }
        redV[t] = best; redI[t] = bi;
        __syncthreads();
        for (int stp = 512; stp > 0; stp >>= 1) {
            if (t < stp) {
                float v2 = redV[t + stp]; int i2 = redI[t + stp];
                if (v2 > redV[t] || (v2 == redV[t] && i2 < redI[t])) {
                    redV[t] = v2; redI[t] = i2;
                }
            }
            __syncthreads();
        }
        if (t == 0) {
            const int oi = redI[0];
            const float os = redV[0];
            keptS[oi] = -1e30f;
            float* ob = outBoxes + ((size_t)b * OUT_K + k) * 4;
            ob[0] = topBox[oi * 4 + 0]; ob[1] = topBox[oi * 4 + 1];
            ob[2] = topBox[oi * 4 + 2]; ob[3] = topBox[oi * 4 + 3];
            outCats  [b * OUT_K + k] = (float)topCat[oi];
            outScores[b * OUT_K + k] = os;
            outValid [b * OUT_K + k] = (os > 0.0f) ? 1.0f : 0.0f;
        }
        __syncthreads();
    }
}

// ---------------------------------------------------------------------------
// Launch
// ---------------------------------------------------------------------------
extern "C" void kernel_launch(void* const* d_in, const int* in_sizes, int n_in,
                              void* d_out, int out_size, void* d_ws, size_t ws_size,
                              hipStream_t stream) {
    const float* feats[5] = {(const float*)d_in[0], (const float*)d_in[1],
                             (const float*)d_in[2], (const float*)d_in[3],
                             (const float*)d_in[4]};
    const float* reg_w   = (const float*)d_in[5];
    const float* reg_b   = (const float*)d_in[6];
    const float* cls_w   = (const float*)d_in[7];
    const float* cls_b   = (const float*)d_in[8];
    const float* anchors = (const float*)d_in[9];

    // workspace layout (all offsets 256B aligned)
    char* ws = (char*)d_ws;
    const size_t OFF_FEAT  = 0;                                  // bf16 16*PAD_PER_IMG
    const size_t OFF_WT    = OFF_FEAT + (size_t)BIMG * PAD_PER_IMG * 2;    // 48,922,624
    const size_t OFF_DELTA = OFF_WT   + (size_t)NLEV * WROWS * KTOT * 2;   // +2,211,840
    const size_t OFF_CONF  = OFF_DELTA + (size_t)BIMG * NANCH * 4 * 4;     // +8,380,416
    const size_t OFF_BOX   = OFF_CONF  + (size_t)BIMG * NANCH * 9 * 4;     // +18,855,936
    const size_t OFF_S     = OFF_BOX   + (size_t)BIMG * NANCH * 4 * 4;
    const size_t OFF_CAT   = OFF_S     + (size_t)BIMG * NANCH * 4;

    unsigned short* featB = (unsigned short*)(ws + OFF_FEAT);
    unsigned short* Wt    = (unsigned short*)(ws + OFF_WT);
    float* delta = (float*)(ws + OFF_DELTA);
    float* conf  = (float*)(ws + OFF_CONF);
    float* boxes = (float*)(ws + OFF_BOX);
    float* sArr  = (float*)(ws + OFF_S);
    int*   catA  = (int*)  (ws + OFF_CAT);

    const int padTotal = BIMG * PAD_PER_IMG;                     // 24,461,312
    zero_feat<<<(padTotal + 255) / 256, 256, 0, stream>>>(featB, padTotal);

    prepack_weights<<<(NLEV * WROWS * KTOT + 255) / 256, 256, 0, stream>>>(reg_w, cls_w, Wt);

    const int sv[5]     = {64, 32, 16, 8, 4};
    const int padoff[5] = {0, 1115136, 1411072, 1494016, 1519616};
    for (int l = 0; l < 5; ++l) {
        int total = BIMG * CIN * sv[l] * sv[l];
        convert_feat<<<(total + 255) / 256, 256, 0, stream>>>(feats[l], featB + padoff[l],
                                                              sv[l], total);
    }

    const int totalWaves = BIMG * MT_PER_IMG;                    // 5456
    conv_gemm<<<totalWaves / 8, 256, 0, stream>>>(featB, Wt, reg_b, cls_b, delta, conf);

    decode_kernel<<<(BIMG * NANCH + 255) / 256, 256, 0, stream>>>(delta, conf, anchors,
                                                                  boxes, sArr, catA);

    nms_kernel<<<BIMG, 1024, 0, stream>>>(boxes, sArr, catA, (float*)d_out);
}
